// GraphScoreCompute_31928786878552
// MI455X (gfx1250) — matrix operand order repaired and checked by reading, not compiled, not run on hardware
//
#include <hip/hip_runtime.h>
#include <hip/hip_bf16.h>
#include <math.h>

// Problem constants (from reference)
#define Bn    8
#define C1n   128
#define Hn    160
#define Wn    160
#define HWn   (Hn*Wn)      // 25600
#define NHn   4
#define HCn   32
#define NTXTn 32
#define GCn   512
#define ECn   128
#define NSEL  64

typedef __attribute__((ext_vector_type(16))) __bf16 v16bf;
typedef __attribute__((ext_vector_type(8)))  float  v8f;

__device__ __forceinline__ unsigned short f2bf(float f) {
  union { float f; unsigned u; } x; x.f = f;
  unsigned r = x.u + 0x7FFFu + ((x.u >> 16) & 1u);   // round-to-nearest-even
  return (unsigned short)(r >> 16);
}
__device__ __forceinline__ float bf2f(unsigned short u) {
  union { unsigned u; float f; } x; x.u = ((unsigned)u) << 16;
  return x.f;
}
__device__ __forceinline__ float sigmoidf(float z) { return 1.f / (1.f + __expf(-z)); }

// ---------------------------------------------------------------------------
// K0: fold BN (scale/bias) + repack conv weights fp32 [co][ci][3][3] ->
//     bf16 [tap(9)][ci_blk(4)][co(128)][kc(32)]  (kc-contiguous pairs for WMMA A)
// ---------------------------------------------------------------------------
__global__ __launch_bounds__(256) void k0_prep(
    const float* __restrict__ wproj, const float* __restrict__ gamma,
    const float* __restrict__ beta,  const float* __restrict__ mean,
    const float* __restrict__ var,   unsigned short* __restrict__ wbf,
    float* __restrict__ scale, float* __restrict__ bias2) {
  int idx = blockIdx.x * 256 + threadIdx.x;
  if (idx < C1n) {
    float s = gamma[idx] * rsqrtf(var[idx] + 1e-5f);
    scale[idx] = s;
    bias2[idx] = beta[idx] - mean[idx] * s;
  }
  if (idx < 128 * 128 * 9) {
    int s  = idx % 3;
    int r  = (idx / 3) % 3;
    int ci = (idx / 9) % 128;
    int co = idx / 1152;
    int tap = r * 3 + s, cib = ci >> 5, kc = ci & 31;
    wbf[((size_t)((tap * 4 + cib) * 128 + co)) * 32 + kc] = f2bf(wproj[idx]);
  }
}

// ---------------------------------------------------------------------------
// K1: spatial modulation head-mean sigmoid + pixel importance (mean |x| over C)
// ---------------------------------------------------------------------------
__global__ __launch_bounds__(256) void k1_spatial(
    const float* __restrict__ x,  const float* __restrict__ w1,
    const float* __restrict__ b1, const float* __restrict__ w2,
    const float* __restrict__ b2, float* __restrict__ sm_mean,
    float* __restrict__ imp) {
  int idx = blockIdx.x * 256 + threadIdx.x;         // (b, hw)
  if (idx >= Bn * HWn) return;
  int b = idx / HWn, hw = idx - b * HWn;
  const float* xp = x + (size_t)b * C1n * HWn + hw;
  float h1[32];
  #pragma unroll
  for (int o = 0; o < 32; ++o) h1[o] = b1[o];
  float ai = 0.f;
  for (int c = 0; c < 128; ++c) {
    float xv = xp[(size_t)c * HWn];
    ai += fabsf(xv);
    #pragma unroll
    for (int o = 0; o < 32; ++o) h1[o] = fmaf(w1[o * 128 + c], xv, h1[o]);
  }
  imp[idx] = ai * (1.f / 128.f);
  #pragma unroll
  for (int o = 0; o < 32; ++o) h1[o] = fmaxf(h1[o], 0.f);
  float sm = 0.f;
  #pragma unroll
  for (int o2 = 0; o2 < 4; ++o2) {
    float a = b2[o2];
    #pragma unroll
    for (int o = 0; o < 32; ++o) a = fmaf(w2[o2 * 32 + o], h1[o], a);
    sm += sigmoidf(a);
  }
  sm_mean[idx] = sm * 0.25f;
}

// ---------------------------------------------------------------------------
// K2: per-image top-64 (iterated block argmax, first-occurrence tie-break),
//     scatter the analytically-exact pixel weight 1/64 into gmw.
//     (softmax rows sum to 1  =>  attn.mean(axis=(0,2)) == 1/64 identically,
//      so the whole KNN/qkv/edge-MLP graph-attention block is dead compute.)
// ---------------------------------------------------------------------------
__global__ __launch_bounds__(256) void k2_topk(float* __restrict__ imp,
                                               float* __restrict__ gmw) {
  int b = blockIdx.x, tid = threadIdx.x;
  float* ip = imp + (size_t)b * HWn;
  float* gp = gmw + (size_t)b * HWn;
  for (int i = tid; i < HWn; i += 256) gp[i] = 0.f;
  __syncthreads();
  __shared__ float sv[256];
  __shared__ int   si[256];
  for (int iter = 0; iter < NSEL; ++iter) {
    float best = -INFINITY; int bi = 0x7fffffff;
    for (int i = tid; i < HWn; i += 256) {
      float v = ip[i];
      if (v > best) { best = v; bi = i; }
    }
    sv[tid] = best; si[tid] = bi;
    __syncthreads();
    for (int s = 128; s > 0; s >>= 1) {
      if (tid < s) {
        float ov = sv[tid + s]; int oi = si[tid + s];
        if (ov > sv[tid] || (ov == sv[tid] && oi < si[tid])) { sv[tid] = ov; si[tid] = oi; }
      }
      __syncthreads();
    }
    if (tid == 0) {
      int wsel = si[0];
      ip[wsel] = -INFINITY;
      gp[wsel] = 0.015625f;   // == 1/64 exactly
    }
    __syncthreads();
  }
}

// ---------------------------------------------------------------------------
// K4: x_new = x * (1 + 0.1*(0.7*sm_mean + 0.3*gmw)), cast to bf16
// ---------------------------------------------------------------------------
__global__ __launch_bounds__(256) void k4_modcast(
    const float* __restrict__ x, const float* __restrict__ sm_mean,
    const float* __restrict__ gmw, unsigned short* __restrict__ xbf) {
  size_t i = (size_t)blockIdx.x * 256 + threadIdx.x;
  if (i >= (size_t)Bn * C1n * HWn) return;
  int hw = (int)(i % HWn);
  int b  = (int)(i / ((size_t)C1n * HWn));
  float m = 1.f + 0.1f * (0.7f * sm_mean[b * HWn + hw] + 0.3f * gmw[b * HWn + hw]);
  xbf[i] = f2bf(x[i] * m);
}

// ---------------------------------------------------------------------------
// K5a: g = guide @ w_gl^T + b_gl   -> [B][32 tokens][128]
// ---------------------------------------------------------------------------
__global__ __launch_bounds__(256) void k5a_guide(
    const float* __restrict__ guide, const float* __restrict__ wgl,
    const float* __restrict__ bgl, float* __restrict__ g) {
  int b = blockIdx.x;
  for (int o = threadIdx.x; o < NTXTn * ECn; o += 256) {
    int n = o >> 7, ec = o & 127;
    const float* gu = guide + ((size_t)b * NTXTn + n) * GCn;
    const float* wr = wgl + (size_t)ec * GCn;
    float acc = bgl[ec];
    for (int k = 0; k < GCn; ++k) acc = fmaf(gu[k], wr[k], acc);
    g[((size_t)b * NTXTn + n) * ECn + ec] = acc;
  }
}

// ---------------------------------------------------------------------------
// K5b: aw[b,m,hw] = sigmoid(max_n <embed, g[b,n,m,:]> / sqrt(32) + bias[m])
// ---------------------------------------------------------------------------
__global__ __launch_bounds__(256) void k5b_aw(
    const unsigned short* __restrict__ xbf, const float* __restrict__ g,
    const float* __restrict__ abias, float* __restrict__ aw) {
  int hw = blockIdx.x * 256 + threadIdx.x;            // 100 blocks * 256 = 25600
  int bm = blockIdx.y; int b = bm >> 2, m = bm & 3;
  float e[32];
  const unsigned short* ep = xbf + ((size_t)b * C1n + m * HCn) * HWn + hw;
  #pragma unroll
  for (int j = 0; j < 32; ++j) e[j] = bf2f(ep[(size_t)j * HWn]);
  float mx = -INFINITY;
  for (int n = 0; n < NTXTn; ++n) {
    const float* gp = g + ((size_t)(b * NTXTn + n)) * ECn + m * HCn;
    float d = 0.f;
    #pragma unroll
    for (int j = 0; j < 32; ++j) d = fmaf(e[j], gp[j], d);
    mx = fmaxf(mx, d);
  }
  aw[(size_t)bm * HWn + hw] = sigmoidf(mx * 0.17677669529663687f + abias[m]);
}

// ---------------------------------------------------------------------------
// K5c: aw_single = mean over heads
// ---------------------------------------------------------------------------
__global__ __launch_bounds__(256) void k5c_awmean(const float* __restrict__ aw,
                                                  float* __restrict__ out2) {
  int idx = blockIdx.x * 256 + threadIdx.x;
  if (idx >= Bn * HWn) return;
  int b = idx / HWn, hw = idx - b * HWn;
  const float* a = aw + (size_t)b * NHn * HWn + hw;
  out2[idx] = 0.25f * (a[0] + a[HWn] + a[2 * HWn] + a[3 * HWn]);
}

// ---------------------------------------------------------------------------
// K6: 3x3 conv as implicit GEMM with v_wmma_f32_16x16x32_bf16.
//   Block = (b, h, 32-pixel strip). M=128 co, N=32 px, K=1152 (9 taps x 128 ci).
//   LDS window [3 rows][34 w][128 ci] bf16: ci-contiguous so each lane's B
//   fragment (K = elem + 16*(lane>=16), N = lane&15) is one contiguous 32B read.
//   A fragments straight from L2-resident repacked weights: two 16B loads per
//   lane in the same k order as B (K = e + 16*(lane>=16)).
//   Epilogue: BN scale/bias + per-head aw gate using the f32 C/D layout
//   (M = vgpr + 8*(lane>=16), N = lane&15).
// ---------------------------------------------------------------------------
__global__ __launch_bounds__(256) void k6_conv(
    const unsigned short* __restrict__ xbf, const unsigned short* __restrict__ wbf,
    const float* __restrict__ scale, const float* __restrict__ bias2,
    const float* __restrict__ aw, float* __restrict__ out) {
  __shared__ __align__(16) unsigned short win[3][34][136];  // pad 128->136
  const int b = blockIdx.z, h = blockIdx.y, w0 = blockIdx.x * 32;
  const int tid = threadIdx.x;

  // Stage halo window once (coalesced across wl at fixed ci).
  for (int idx = tid; idx < 3 * 34; idx += 256) {
    int r = idx / 34, wl = idx - r * 34;
    int hh = h + r - 1, ww = w0 + wl - 1;
    if (hh >= 0 && hh < Hn && ww >= 0 && ww < Wn) {
      const unsigned short* src = xbf + (size_t)b * C1n * HWn + (size_t)hh * Wn + ww;
      for (int ci = 0; ci < C1n; ++ci) win[r][wl][ci] = src[(size_t)ci * HWn];
    } else {
      for (int ci = 0; ci < C1n; ++ci) win[r][wl][ci] = 0;
    }
  }
  __syncthreads();

  const int wave = tid >> 5, lane = tid & 31;
  const int mrow = lane & 15, hi = lane >> 4;

  union Frag { v16bf v; float4 f4[2]; };
  v8f acc0 = {0.f, 0.f, 0.f, 0.f, 0.f, 0.f, 0.f, 0.f};
  v8f acc1 = acc0;

  const unsigned short* wbase = wbf + ((size_t)(wave * 16 + mrow)) * 32 + hi * 16;

  #pragma unroll 1
  for (int tap = 0; tap < 9; ++tap) {
    const int r = tap / 3, s = tap - r * 3;
    #pragma unroll
    for (int cib = 0; cib < 4; ++cib) {
      Frag a, b0f, b1f;
      const float4* ap = (const float4*)(const void*)(wbase + ((size_t)(tap * 4 + cib)) * (128 * 32));
      a.f4[0] = ap[0];         // A elems 0-7  : K = hi*16 .. hi*16+7
      a.f4[1] = ap[1];         // A elems 8-15 : K = hi*16+8 .. hi*16+15
      const unsigned short* bp0 = &win[r][mrow + s][cib * 32 + hi * 16];
      b0f.f4[0] = *(const float4*)(const void*)bp0;
      b0f.f4[1] = *(const float4*)(const void*)(bp0 + 8);
      const unsigned short* bp1 = &win[r][mrow + 16 + s][cib * 32 + hi * 16];
      b1f.f4[0] = *(const float4*)(const void*)bp1;
      b1f.f4[1] = *(const float4*)(const void*)(bp1 + 8);
      acc0 = __builtin_amdgcn_wmma_f32_16x16x32_bf16(false, a.v, false, b0f.v,
                                                     (short)0, acc0, false, false);
      acc1 = __builtin_amdgcn_wmma_f32_16x16x32_bf16(false, a.v, false, b1f.v,
                                                     (short)0, acc1, false, false);
    }
  }

  const int px = h * Wn + w0 + mrow;
  #pragma unroll
  for (int rr = 0; rr < 8; ++rr) {
    int co = wave * 16 + rr + 8 * hi;
    float sc = scale[co], bs = bias2[co];
    size_t abase = ((size_t)b * NHn + (co >> 5)) * HWn + px;
    size_t obase = ((size_t)b * C1n + co) * HWn + px;
    out[obase]      = fmaf(acc0[rr], sc, bs) * aw[abase];
    out[obase + 16] = fmaf(acc1[rr], sc, bs) * aw[abase + 16];
  }
}

// ---------------------------------------------------------------------------
// Launch.  Workspace need: ~59 MB
//   xbf 52.4MB | wbf 0.3MB | sm/imp/gmw 3x0.82MB | aw 3.28MB | g 0.13MB | bn 1KB
// ---------------------------------------------------------------------------
extern "C" void kernel_launch(void* const* d_in, const int* in_sizes, int n_in,
                              void* d_out, int out_size, void* d_ws, size_t ws_size,
                              hipStream_t stream) {
  (void)in_sizes; (void)n_in; (void)out_size; (void)ws_size;
  const float* x      = (const float*)d_in[0];
  const float* guide  = (const float*)d_in[1];
  const float* w_gm1  = (const float*)d_in[2];
  const float* b_gm1  = (const float*)d_in[3];
  const float* w_gm2  = (const float*)d_in[4];
  const float* b_gm2  = (const float*)d_in[5];
  // d_in[6..10] (w_qkv, w_ge1, b_ge1, w_ge2, b_ge2): mathematically dead —
  // attn.mean over (heads, keys) of a row-softmax is identically 1/64.
  const float* w_gl   = (const float*)d_in[11];
  const float* b_gl   = (const float*)d_in[12];
  const float* abias  = (const float*)d_in[13];
  const float* w_proj = (const float*)d_in[14];
  const float* gamma  = (const float*)d_in[15];
  const float* beta   = (const float*)d_in[16];
  const float* mean   = (const float*)d_in[17];
  const float* var    = (const float*)d_in[18];

  char* ws = (char*)d_ws;
  size_t off = 0;
  auto take = [&](size_t bytes) -> char* {
    char* p = ws + off;
    off += (bytes + 255) & ~(size_t)255;
    return p;
  };
  unsigned short* xbf = (unsigned short*)take((size_t)Bn * C1n * HWn * 2);
  unsigned short* wbf = (unsigned short*)take((size_t)9 * 4 * 128 * 32 * 2);
  float* sm  = (float*)take((size_t)Bn * HWn * 4);
  float* imp = (float*)take((size_t)Bn * HWn * 4);
  float* gmw = (float*)take((size_t)Bn * HWn * 4);
  float* awb = (float*)take((size_t)Bn * NHn * HWn * 4);
  float* gb  = (float*)take((size_t)Bn * NTXTn * ECn * 4);
  float* scl = (float*)take(128 * 4);
  float* bs2 = (float*)take(128 * 4);

  float* out    = (float*)d_out;
  float* out_aw = out + (size_t)Bn * C1n * HWn;

  k0_prep<<<(128 * 128 * 9 + 255) / 256, 256, 0, stream>>>(
      w_proj, gamma, beta, mean, var, wbf, scl, bs2);
  k1_spatial<<<(Bn * HWn + 255) / 256, 256, 0, stream>>>(
      x, w_gm1, b_gm1, w_gm2, b_gm2, sm, imp);
  k2_topk<<<Bn, 256, 0, stream>>>(imp, gmw);
  k4_modcast<<<(int)(((size_t)Bn * C1n * HWn + 255) / 256), 256, 0, stream>>>(
      x, sm, gmw, xbf);
  k5a_guide<<<Bn, 256, 0, stream>>>(guide, w_gl, b_gl, gb);
  k5b_aw<<<dim3(HWn / 256, Bn * NHn), 256, 0, stream>>>(xbf, gb, abias, awb);
  k5c_awmean<<<(Bn * HWn + 255) / 256, 256, 0, stream>>>(awb, out_aw);
  k6_conv<<<dim3(Wn / 32, Hn, Bn), 256, 0, stream>>>(xbf, wbf, scl, bs2, awb, out);
}
